// LiquidCfCExpertRouter_51531017617702
// MI455X (gfx1250) — matrix operand order repaired and checked by reading, not compiled, not run on hardware
//
#include <hip/hip_runtime.h>
#include <hip/hip_bf16.h>
#include <math.h>

// ---------------------------------------------------------------------------
// LiquidCfC expert router, fused for MI455X (gfx1250, wave32, WMMA).
//
// bf16 A/B fragments, f32 accumulation (v_wmma_f32_16x16x32_bf16).
// Weights pre-transposed+converted to bf16 [N][K] in d_ws.
// Stage-1 (x @ Wp, K=4096) uses block-cooperative double-buffered LDS staging
// of the Wp k-slice so 8 waves share one copy instead of 8 redundant L2 reads.
// ---------------------------------------------------------------------------

typedef __bf16 bf16_t;
typedef __attribute__((ext_vector_type(16))) __bf16 v16bf;
typedef __attribute__((ext_vector_type(8)))  __bf16 v8bf;
typedef __attribute__((ext_vector_type(8)))  float  v8f;

constexpr int Hdim = 4096;   // hidden dim (K of GEMM1)
constexpr int Rdim = 256;    // router dim
constexpr int Edim = 64;     // experts
constexpr float TEMPORAL_W = 0.1f;

#define WMMA_BF16(A, Bf, C) \
  __builtin_amdgcn_wmma_f32_16x16x32_bf16(false, (A), false, (Bf), (short)0, (C), false, false)

// ---- prep: dst[n*K + k] = bf16(src[k*N + n])  (transpose + convert) --------
__global__ void transpose_to_bf16(const float* __restrict__ src,
                                  bf16_t* __restrict__ dst, int K, int N) {
  int idx = blockIdx.x * blockDim.x + threadIdx.x;
  if (idx >= K * N) return;
  int n = idx / K;
  int k = idx - n * K;
  dst[idx] = (bf16_t)src[(size_t)k * N + n];
}

// ---- fragment builders -----------------------------------------------------
// A-frag (16x32 bf16): lane holds row M = lane&15.
//   lanes 0-15 : elems 0..15 -> K=k0+0..15
//   lanes 16-31: elems 0..15 -> K=k0+16..31
__device__ inline v16bf afrag_from_f32(const float* __restrict__ row, int k0, int kbA) {
  const float4* p0 = (const float4*)(row + k0 + kbA);        // 8 floats
  const float4* p1 = (const float4*)(row + k0 + kbA + 8);    // 8 floats
  float4 a = p0[0], b = p0[1], c = p1[0], d = p1[1];
  v16bf f;
  f[0]=(bf16_t)a.x; f[1]=(bf16_t)a.y; f[2]=(bf16_t)a.z; f[3]=(bf16_t)a.w;
  f[4]=(bf16_t)b.x; f[5]=(bf16_t)b.y; f[6]=(bf16_t)b.z; f[7]=(bf16_t)b.w;
  f[8]=(bf16_t)c.x; f[9]=(bf16_t)c.y; f[10]=(bf16_t)c.z; f[11]=(bf16_t)c.w;
  f[12]=(bf16_t)d.x; f[13]=(bf16_t)d.y; f[14]=(bf16_t)d.z; f[15]=(bf16_t)d.w;
  return f;
}

__device__ inline v16bf afrag_from_lds(const bf16_t* __restrict__ slice,
                                       int m, int k0, int kbA) {
  v8bf lo = *(const v8bf*)(slice + m * Rdim + k0 + kbA);
  v8bf hi = *(const v8bf*)(slice + m * Rdim + k0 + kbA + 8);
  v16bf f;
#pragma unroll
  for (int i = 0; i < 8; ++i) { f[i] = lo[i]; f[8 + i] = hi[i]; }
  return f;
}

// B-frag (32x16 bf16): lane holds col N = lane&15; lanes 0-15 K=k0+0..15,
// lanes 16-31 K=k0+16..31 (contiguous 32B per lane).
__device__ inline v16bf bfrag_global(const bf16_t* __restrict__ wT, int Kstride,
                                     int col, int k0, int kbB) {
  return *(const v16bf*)(wT + (size_t)col * Kstride + k0 + kbB);
}

// Staged k-slice layout: wlds[col*32 + (k - k0)], 32 bf16 per column.
__device__ inline v16bf bfrag_lds(const bf16_t* __restrict__ wlds, int col, int kbB) {
  return *(const v16bf*)(wlds + col * 32 + kbB);
}

// Stage one WpT k-slice (256 cols x 32 K bf16 = 16 KB = 1024 uint4) into LDS.
// 256 threads: 4 uint4 each; coalesced global reads, contiguous LDS stores.
__device__ inline void stage_wp(const bf16_t* __restrict__ WpT, bf16_t* __restrict__ dst,
                                int k0, int tid) {
  uint4* d = (uint4*)dst;
#pragma unroll
  for (int i = 0; i < 4; ++i) {
    int flat = tid + i * 256;        // uint4 index in slab
    int n    = flat >> 2;            // column (0..255)
    int q    = flat & 3;             // 8-element chunk within the 32-K row
    d[flat] = *(const uint4*)(WpT + (size_t)n * Hdim + k0 + q * 8);
  }
}

// ---- fused router kernel ---------------------------------------------------
__global__ __launch_bounds__(256) void liquid_router_kernel(
    const float* __restrict__ x, const float* __restrict__ h,
    const int* __restrict__ prev_sel,
    const float* __restrict__ bp, const float* __restrict__ tau,
    const float* __restrict__ bg,
    const bf16_t* __restrict__ WpT, const bf16_t* __restrict__ AT,
    const bf16_t* __restrict__ BmT, const bf16_t* __restrict__ WgT,
    float* __restrict__ out_idx, float* __restrict__ out_w)
{
  // 64 KB arena, time-shared:
  //   stage-1 loop : [0,16K) wbuf0, [16K,32K) wbuf1   (Wp k-slice double buffer)
  //   afterwards   : 8 waves x 8 KB slices (x_proj -> h_new -> logits)
  __shared__ __align__(16) char smem_raw[65536];
  bf16_t* wbuf0 = (bf16_t*)smem_raw;
  bf16_t* wbuf1 = (bf16_t*)(smem_raw + 16384);

  const int tid   = threadIdx.x;
  const int lane  = tid & 31;
  const int wave  = tid >> 5;
  const int rowBase = blockIdx.x * 128 + wave * 16;   // 16 rows per wave
  bf16_t* slice = (bf16_t*)(smem_raw + wave * 8192);

  const int nIdx  = lane & 15;          // C-layout column / A-frag row
  const int mhalf = lane >> 4;          // C-layout row half select
  const int kbA   = mhalf ? 16 : 0;     // A-frag K base
  const int kbB   = mhalf ? 16 : 0;     // B-frag K base

  // ======== Stage 1: x_proj = x @ Wp   (K = 4096, staged B) ========
  v8f acc[16] = {};
  const float* xrow = x + (size_t)(rowBase + nIdx) * Hdim;

  stage_wp(WpT, wbuf0, 0, tid);
  __syncthreads();
#pragma unroll 1
  for (int k0 = 0; k0 < Hdim; k0 += 32) {
    bf16_t* cur = ((k0 >> 5) & 1) ? wbuf1 : wbuf0;
    bf16_t* nxt = ((k0 >> 5) & 1) ? wbuf0 : wbuf1;
    if (k0 + 32 < Hdim) stage_wp(WpT, nxt, k0 + 32, tid);   // prefetch next slice
    v16bf a = afrag_from_f32(xrow, k0, kbA);
#pragma unroll
    for (int t = 0; t < 16; ++t) {
      v16bf b = bfrag_lds(cur, t * 16 + nIdx, kbB);
      acc[t] = WMMA_BF16(a, b, acc[t]);
    }
    __syncthreads();
  }

  // + bp, stash x_proj (bf16) row-major into this wave's LDS slice
#pragma unroll
  for (int t = 0; t < 16; ++t) {
    float bias = bp[t * 16 + nIdx];
#pragma unroll
    for (int r = 0; r < 8; ++r) {
      int rl = r + 8 * mhalf;
      slice[rl * Rdim + t * 16 + nIdx] = (bf16_t)(acc[t][r] + bias);
    }
  }
  __syncthreads();

  // ======== Stage 2: u = h@A + x_proj@Bm   (K = 256) ========
  v8f acc2[16] = {};
  const float* hrow = h + (size_t)(rowBase + nIdx) * Rdim;
#pragma unroll 1
  for (int k0 = 0; k0 < Rdim; k0 += 32) {
    v16bf ah = afrag_from_f32(hrow, k0, kbA);
    v16bf ax = afrag_from_lds(slice, nIdx, k0, kbA);
#pragma unroll
    for (int t = 0; t < 16; ++t) {
      v16bf b1 = bfrag_global(AT,  Rdim, t * 16 + nIdx, k0, kbB);
      v16bf b2 = bfrag_global(BmT, Rdim, t * 16 + nIdx, k0, kbB);
      acc2[t] = WMMA_BF16(ah, b1, acc2[t]);
      acc2[t] = WMMA_BF16(ax, b2, acc2[t]);
    }
  }
  __syncthreads();

  // ======== Stage 3: CfC update, h_new -> LDS (bf16) ========
#pragma unroll
  for (int t = 0; t < 16; ++t) {
    int col = t * 16 + nIdx;
    float tv = tau[col];
#pragma unroll
    for (int r = 0; r < 8; ++r) {
      int rl = r + 8 * mhalf;
      float hv   = h[(size_t)(rowBase + rl) * Rdim + col];
      float dhdt = -hv / tv + tanhf(acc2[t][r]);
      slice[rl * Rdim + col] = (bf16_t)(hv + 0.1f * dhdt);
    }
  }
  __syncthreads();

  // ======== Stage 4: logits = h_new @ Wg   (K = 256, N = 64) ========
  v8f acc3[4] = {};
#pragma unroll 1
  for (int k0 = 0; k0 < Rdim; k0 += 32) {
    v16bf ah = afrag_from_lds(slice, nIdx, k0, kbA);
#pragma unroll
    for (int t = 0; t < 4; ++t) {
      v16bf b = bfrag_global(WgT, Rdim, t * 16 + nIdx, k0, kbB);
      acc3[t] = WMMA_BF16(ah, b, acc3[t]);
    }
  }
  __syncthreads();  // all h_new reads done before slice is reused for logits

  // ======== Stage 5: + bg, temporal bias; logits (f32) -> LDS ========
  int p0[8], p1[8];
#pragma unroll
  for (int r = 0; r < 8; ++r) {
    int rg = rowBase + r + 8 * mhalf;
    p0[r] = prev_sel[rg * 2 + 0];
    p1[r] = prev_sel[rg * 2 + 1];
  }
  float* lf = (float*)slice;  // 16*64*4 = 4 KB inside the 8 KB slice
#pragma unroll
  for (int t = 0; t < 4; ++t) {
    int e = t * 16 + nIdx;
    float bgv = bg[e];
#pragma unroll
    for (int r = 0; r < 8; ++r) {
      float v = acc3[t][r] + bgv;
      // scatter .set semantics: duplicate prev_sel gets +0.1 once
      if (e == p0[r] || e == p1[r]) v += TEMPORAL_W;
      lf[(r + 8 * mhalf) * Edim + e] = v;
    }
  }
  __syncthreads();

  // ======== Stage 6: top-2 over 64 experts + softmax(2) ========
  if (lane < 16) {
    const float* rowp = lf + lane * Edim;
    float v1 = -3.4e38f, v2 = -3.4e38f;
    int   i1 = 0,        i2 = 0;
#pragma unroll 1
    for (int e = 0; e < Edim; ++e) {
      float v = rowp[e];
      if (v > v1)      { v2 = v1; i2 = i1; v1 = v; i1 = e; }
      else if (v > v2) { v2 = v;  i2 = e; }
    }
    float e2  = __expf(v2 - v1);
    float inv = 1.0f / (1.0f + e2);
    size_t rg = (size_t)rowBase + lane;
    out_idx[rg * 2 + 0] = (float)i1;
    out_idx[rg * 2 + 1] = (float)i2;
    out_w[rg * 2 + 0]   = inv;
    out_w[rg * 2 + 1]   = e2 * inv;
  }
}

// ---------------------------------------------------------------------------
extern "C" void kernel_launch(void* const* d_in, const int* in_sizes, int n_in,
                              void* d_out, int out_size, void* d_ws, size_t ws_size,
                              hipStream_t stream) {
  const float* x        = (const float*)d_in[0];
  const float* h        = (const float*)d_in[1];
  const int*   prev_sel = (const int*)  d_in[2];
  const float* Wp       = (const float*)d_in[3];
  const float* bp       = (const float*)d_in[4];
  const float* tau      = (const float*)d_in[5];
  const float* A        = (const float*)d_in[6];
  const float* Bm       = (const float*)d_in[7];
  const float* Wg       = (const float*)d_in[8];
  const float* bg       = (const float*)d_in[9];

  const int B = in_sizes[1] / Rdim;   // h is [B, R]

  // Workspace: bf16 transposed weights ([N][K] layout).
  bf16_t* WpT = (bf16_t*)d_ws;                       // R*H  = 1M elems (2 MiB)
  bf16_t* AT  = WpT + (size_t)Rdim * Hdim;           // R*R
  bf16_t* BmT = AT  + (size_t)Rdim * Rdim;           // R*R
  bf16_t* WgT = BmT + (size_t)Rdim * Rdim;           // E*R

  const int thr = 256;
  transpose_to_bf16<<<(Rdim * Hdim + thr - 1) / thr, thr, 0, stream>>>(Wp, WpT, Hdim, Rdim);
  transpose_to_bf16<<<(Rdim * Rdim + thr - 1) / thr, thr, 0, stream>>>(A,  AT,  Rdim, Rdim);
  transpose_to_bf16<<<(Rdim * Rdim + thr - 1) / thr, thr, 0, stream>>>(Bm, BmT, Rdim, Rdim);
  transpose_to_bf16<<<(Edim * Rdim + thr - 1) / thr, thr, 0, stream>>>(Wg, WgT, Rdim, Edim);

  // d_out = [top_idx (B*2), top_w (B*2)] flattened in return order.
  float* out_idx = (float*)d_out;
  float* out_w   = out_idx + (size_t)B * 2;

  liquid_router_kernel<<<B / 128, 256, 0, stream>>>(
      x, h, prev_sel, bp, tau, bg, WpT, AT, BmT, WgT, out_idx, out_w);
}
